// MultiHeadAttention_5265629905091
// MI455X (gfx1250) — matrix operand order repaired and checked by reading, not compiled, not run on hardware
//
#include <hip/hip_runtime.h>

#define NHEAD 16
#define HD    64
#define DM    1024
#define BATCH 8
#define SEQ   1024
#define MROWS (BATCH * SEQ) /* 8192 */

typedef __attribute__((ext_vector_type(16))) __bf16 bf16x16;
typedef __attribute__((ext_vector_type(8)))  __bf16 bf16x8;
typedef __attribute__((ext_vector_type(8)))  float  f32x8;

// ---------------------------------------------------------------------------
// WMMA fragment loaders (bf16, 16x16x32), per CDNA5 ISA layouts.
// A (16x32, MxK): lanes 0-15 -> row M=lane, K = {k0..k0+7, k0+16..k0+23};
//                 lanes 16-31 -> row M=lane-16, K = {k0+8.., k0+24..}.
// B (32x16, KxN): lane n%16 holds column N; lane<16 -> K {0..7, 16..23},
//                 lane>=16 -> K {8..15, 24..31} (same K order as A).
// Both loaders read K-contiguous row-major storage, so they implement A @ B^T
// for row-major A[M,K] and row-major B[N,K] (i.e. nn.Linear y = x W^T).
// ---------------------------------------------------------------------------
__device__ __forceinline__ bf16x16 load_fragA(const __bf16* base, int row0, int k0,
                                              int ld, int lane) {
  int r   = row0 + (lane & 15);
  int klo = k0 + ((lane >> 4) << 3);
  const __bf16* p = base + (size_t)r * ld + klo;
  bf16x8 lo = *(const bf16x8*)p;        // K = klo .. klo+7
  bf16x8 hi = *(const bf16x8*)(p + 16); // K = klo+16 .. klo+23
  bf16x16 out;
#pragma unroll
  for (int i = 0; i < 8; i++) { out[i] = lo[i]; out[i + 8] = hi[i]; }
  return out;
}

__device__ __forceinline__ bf16x16 load_fragB(const __bf16* base, int n0, int k0,
                                              int ld, int lane) {
  int n = n0 + (lane & 15);
  int k = k0 + ((lane >> 4) << 3);
  const __bf16* p = base + (size_t)n * ld + k;
  bf16x8 lo = *(const bf16x8*)p;       // K = k .. k+7
  bf16x8 hi = *(const bf16x8*)(p + 16); // K = k+16 .. k+23
  bf16x16 out;
#pragma unroll
  for (int i = 0; i < 8; i++) { out[i] = lo[i]; out[i + 8] = hi[i]; }
  return out;
}

__device__ __forceinline__ f32x8 wmma_bf16(bf16x16 a, bf16x16 b, f32x8 c) {
  return __builtin_amdgcn_wmma_f32_16x16x32_bf16(
      /*neg_a=*/false, a, /*neg_b=*/false, b,
      /*c_mod=*/(short)0, c, /*reuse_a=*/false, /*reuse_b=*/false);
}

// ---------------------------------------------------------------------------
// f32 -> bf16 down-conversion (bandwidth-trivial, done once per tensor)
// ---------------------------------------------------------------------------
__global__ void cvt_f32_bf16(const float* __restrict__ in, __bf16* __restrict__ out,
                             int n) {
  int i = blockIdx.x * blockDim.x + threadIdx.x;
  int stride = gridDim.x * blockDim.x;
  for (; i < n; i += stride) out[i] = (__bf16)in[i];
}

// ---------------------------------------------------------------------------
// C = A @ B^T, A row-major [M,K] bf16, B row-major [N,K] bf16.
// 256 threads = 8 waves; block tile 128x128; wave tile 64x32 (4x2 WMMA accums).
// ---------------------------------------------------------------------------
template <bool OUT_BF16>
__global__ __launch_bounds__(256) void gemm_abt(const __bf16* __restrict__ A,
                                                const __bf16* __restrict__ Bm,
                                                void* __restrict__ Cout,
                                                int M, int N, int K) {
  int lane = threadIdx.x & 31;
  int wave = threadIdx.x >> 5;
  int row0 = blockIdx.y * 128 + (wave >> 2) * 64;
  int col0 = blockIdx.x * 128 + (wave & 3) * 32;
  f32x8 acc[4][2] = {};
  for (int kk = 0; kk < K; kk += 32) {
    bf16x16 a[4], b[2];
#pragma unroll
    for (int mt = 0; mt < 4; mt++) a[mt] = load_fragA(A, row0 + mt * 16, kk, K, lane);
#pragma unroll
    for (int nt = 0; nt < 2; nt++) b[nt] = load_fragB(Bm, col0 + nt * 16, kk, K, lane);
#pragma unroll
    for (int mt = 0; mt < 4; mt++)
#pragma unroll
      for (int nt = 0; nt < 2; nt++)
        acc[mt][nt] = wmma_bf16(a[mt], b[nt], acc[mt][nt]);
  }
  int mh = (lane >> 4) * 8, nl = lane & 15;
#pragma unroll
  for (int mt = 0; mt < 4; mt++)
#pragma unroll
    for (int nt = 0; nt < 2; nt++)
#pragma unroll
      for (int i = 0; i < 8; i++) {
        int m = row0 + mt * 16 + mh + i;
        int n = col0 + nt * 16 + nl;
        float v = acc[mt][nt][i];
        if (OUT_BF16) ((__bf16*)Cout)[(size_t)m * N + n] = (__bf16)v;
        else          ((float*)Cout)[(size_t)m * N + n] = v;
      }
}

// ---------------------------------------------------------------------------
// Per-head q/k/v projection (K=64). blockIdx = (m-block, head, {q,k,v}).
// v is stored TRANSPOSED as vT[h][b][d][t] so the PV GEMM later is also A@B^T.
// ---------------------------------------------------------------------------
__global__ __launch_bounds__(256) void qkv_kernel(const __bf16* __restrict__ xp,
                                                  const __bf16* __restrict__ wq,
                                                  const __bf16* __restrict__ wk,
                                                  const __bf16* __restrict__ wv,
                                                  __bf16* __restrict__ qb,
                                                  __bf16* __restrict__ kb,
                                                  __bf16* __restrict__ vtb) {
  int lane = threadIdx.x & 31, wave = threadIdx.x >> 5;
  int h = blockIdx.y, sel = blockIdx.z;
  const __bf16* W = (sel == 0 ? wq : sel == 1 ? wk : wv) + (size_t)h * HD * HD;
  int row0 = blockIdx.x * 256 + wave * 32;
  bf16x16 a[2][2];
#pragma unroll
  for (int mt = 0; mt < 2; mt++)
#pragma unroll
    for (int ks = 0; ks < 2; ks++)
      a[mt][ks] = load_fragA(xp, row0 + mt * 16, h * HD + ks * 32, DM, lane);
  f32x8 acc[2][4] = {};
#pragma unroll
  for (int ks = 0; ks < 2; ks++)
#pragma unroll
    for (int nt = 0; nt < 4; nt++) {
      bf16x16 bb = load_fragB(W, nt * 16, ks * 32, HD, lane);
#pragma unroll
      for (int mt = 0; mt < 2; mt++) acc[mt][nt] = wmma_bf16(a[mt][ks], bb, acc[mt][nt]);
    }
  int mh = (lane >> 4) * 8, nl = lane & 15;
  if (sel < 2) {
    __bf16* out = (sel == 0 ? qb : kb) + (size_t)h * MROWS * HD;
#pragma unroll
    for (int mt = 0; mt < 2; mt++)
#pragma unroll
      for (int nt = 0; nt < 4; nt++)
#pragma unroll
        for (int i = 0; i < 8; i++) {
          int m = row0 + mt * 16 + mh + i, n = nt * 16 + nl;
          out[(size_t)m * HD + n] = (__bf16)acc[mt][nt][i];
        }
  } else {
#pragma unroll
    for (int mt = 0; mt < 2; mt++)
#pragma unroll
      for (int nt = 0; nt < 4; nt++)
#pragma unroll
        for (int i = 0; i < 8; i++) {
          int m = row0 + mt * 16 + mh + i, n = nt * 16 + nl;
          int bidx = m >> 10, t = m & (SEQ - 1);
          vtb[(((size_t)h * BATCH + bidx) * HD + n) * SEQ + t] = (__bf16)acc[mt][nt][i];
        }
  }
}

// ---------------------------------------------------------------------------
// Attention core: one workgroup (4 waves) per (b, h, 64-row s-block).
// Phase 1: S = q k^T / 8 -> bf16 tile [64][1024] in 128 KB dynamic LDS
//          (CDNA5's 320 KB/WGP LDS makes the whole score row-block resident).
// Phase 2: row max + exp with VECTOR (bf16x8) LDS accesses; the 1/rowsum is
//          folded into the PV epilogue so no normalization pass over LDS.
// Phase 3: out = P @ v via WMMA, A-fragments read straight from LDS.
// ---------------------------------------------------------------------------
__global__ __launch_bounds__(128) void attn_kernel(const __bf16* __restrict__ qb,
                                                   const __bf16* __restrict__ kb,
                                                   const __bf16* __restrict__ vtb,
                                                   __bf16* __restrict__ concat) {
  extern __shared__ __align__(16) char smem_raw[];
  __bf16* P = (__bf16*)smem_raw; // [64][SEQ] bf16 = 128 KB
  __shared__ float red[128];
  __shared__ float rowinv[64];
  int lane = threadIdx.x & 31, wave = threadIdx.x >> 5;
  int sb = blockIdx.x, h = blockIdx.y, b = blockIdx.z;
  int rowbase = b * SEQ + sb * 64;
  const __bf16* qh = qb + (size_t)h * MROWS * HD;
  const __bf16* kh = kb + (size_t)h * MROWS * HD;
  int mh = (lane >> 4) * 8, nl = lane & 15;

  // Phase 1: scores. Each wave covers a 256-wide column strip of all 64 rows.
  bf16x16 a[4][2];
#pragma unroll
  for (int mt = 0; mt < 4; mt++)
#pragma unroll
    for (int ks = 0; ks < 2; ks++)
      a[mt][ks] = load_fragA(qh, rowbase + mt * 16, ks * 32, HD, lane);
  for (int nt = 0; nt < 16; nt++) {
    int t0 = wave * 256 + nt * 16;
    f32x8 acc[4] = {};
#pragma unroll
    for (int ks = 0; ks < 2; ks++) {
      bf16x16 bb = load_fragB(kh, b * SEQ + t0, ks * 32, HD, lane);
#pragma unroll
      for (int mt = 0; mt < 4; mt++) acc[mt] = wmma_bf16(a[mt][ks], bb, acc[mt]);
    }
#pragma unroll
    for (int mt = 0; mt < 4; mt++)
#pragma unroll
      for (int i = 0; i < 8; i++) {
        int m = mt * 16 + mh + i;
        P[(size_t)m * SEQ + t0 + nl] = (__bf16)(acc[mt][i] * 0.125f); // 1/sqrt(64)
      }
  }
  __syncthreads();

  // Phase 2: softmax. Two threads per row; each half-row is 1 KB-aligned, so
  // all LDS traffic is 16-byte bf16x8 vectors (8x fewer ds ops than scalar).
  {
    int tid = threadIdx.x;
    int row = tid >> 1, half = tid & 1;
    bf16x8* pv = (bf16x8*)(P + (size_t)row * SEQ + half * 512);
    float mx = -3.0e38f;
    for (int i = 0; i < 64; i++) {
      bf16x8 v = pv[i];
#pragma unroll
      for (int j = 0; j < 8; j++) mx = fmaxf(mx, (float)v[j]);
    }
    red[tid] = mx;
    __syncthreads();
    float m = fmaxf(red[2 * row], red[2 * row + 1]);
    __syncthreads();
    float sum = 0.f;
    for (int i = 0; i < 64; i++) {
      bf16x8 v = pv[i];
      bf16x8 e;
#pragma unroll
      for (int j = 0; j < 8; j++) {
        float ex = __expf((float)v[j] - m);
        sum += ex;
        e[j] = (__bf16)ex;
      }
      pv[i] = e;
    }
    red[tid] = sum;
    __syncthreads();
    if (half == 0) rowinv[row] = 1.0f / (red[2 * row] + red[2 * row + 1]);
  }
  __syncthreads();

  // Phase 3: out = P @ v. Wave owns one 16-row m-tile; K loops over SEQ.
  const __bf16* vh = vtb + ((size_t)h * BATCH + b) * HD * SEQ;
  f32x8 acc[4] = {};
  int mt = wave;
  for (int kk = 0; kk < SEQ; kk += 32) {
    bf16x16 af = load_fragA(P, mt * 16, kk, SEQ, lane);
#pragma unroll
    for (int nt = 0; nt < 4; nt++) {
      bf16x16 bb = load_fragB(vh, nt * 16, kk, SEQ, lane);
      acc[nt] = wmma_bf16(af, bb, acc[nt]);
    }
  }
#pragma unroll
  for (int nt = 0; nt < 4; nt++)
#pragma unroll
    for (int i = 0; i < 8; i++) {
      int ml = mt * 16 + mh + i;
      int d = nt * 16 + nl;
      concat[(size_t)(rowbase + ml) * DM + h * HD + d] =
          (__bf16)(acc[nt][i] * rowinv[ml]);
    }
}

// ---------------------------------------------------------------------------
// Host-side orchestration. Workspace layout (bf16 elements, ~88.5 MB total);
// xbf is dead after GEMM1 and is reused as the concat buffer.
// ---------------------------------------------------------------------------
extern "C" void kernel_launch(void* const* d_in, const int* in_sizes, int n_in,
                              void* d_out, int out_size, void* d_ws, size_t ws_size,
                              hipStream_t stream) {
  (void)in_sizes; (void)n_in; (void)out_size; (void)ws_size;
  const float* x      = (const float*)d_in[0];
  const float* Wsplit = (const float*)d_in[1];
  const float* Wout   = (const float*)d_in[2];
  const float* Wq     = (const float*)d_in[3];
  const float* Wk     = (const float*)d_in[4];
  const float* Wv     = (const float*)d_in[5];

  __bf16* ws = (__bf16*)d_ws;
  const size_t NX = (size_t)MROWS * DM; // 8388608 elements
  __bf16* xbf       = ws;               // later reused as concat
  __bf16* xp        = ws + 1 * NX;
  __bf16* qbuf      = ws + 2 * NX;
  __bf16* kbuf      = ws + 3 * NX;
  __bf16* vtbuf     = ws + 4 * NX;
  __bf16* wsplit_bf = ws + 5 * NX;
  __bf16* wout_bf   = wsplit_bf + (size_t)DM * DM;
  __bf16* wq_bf     = wout_bf + (size_t)DM * DM;
  __bf16* wk_bf     = wq_bf + NHEAD * HD * HD;
  __bf16* wv_bf     = wk_bf + NHEAD * HD * HD;

  cvt_f32_bf16<<<512, 256, 0, stream>>>(x, xbf, MROWS * DM);
  cvt_f32_bf16<<<256, 256, 0, stream>>>(Wsplit, wsplit_bf, DM * DM);
  cvt_f32_bf16<<<256, 256, 0, stream>>>(Wout, wout_bf, DM * DM);
  cvt_f32_bf16<<<64, 256, 0, stream>>>(Wq, wq_bf, NHEAD * HD * HD);
  cvt_f32_bf16<<<64, 256, 0, stream>>>(Wk, wk_bf, NHEAD * HD * HD);
  cvt_f32_bf16<<<64, 256, 0, stream>>>(Wv, wv_bf, NHEAD * HD * HD);

  // xp = x @ W_split^T
  gemm_abt<true><<<dim3(DM / 128, MROWS / 128), 256, 0, stream>>>(
      xbf, wsplit_bf, xp, MROWS, DM, DM);
  // per-head q, k, v (v transposed)
  qkv_kernel<<<dim3(MROWS / 256, NHEAD, 3), 256, 0, stream>>>(
      xp, wq_bf, wk_bf, wv_bf, qbuf, kbuf, vtbuf);
  // attention core -> concat (reusing xbf)
  attn_kernel<<<dim3(SEQ / 64, NHEAD, BATCH), 128,
                64 * SEQ * sizeof(__bf16), stream>>>(qbuf, kbuf, vtbuf, xbf);
  // y = concat @ W_out^T (f32 output)
  gemm_abt<false><<<dim3(DM / 128, MROWS / 128), 256, 0, stream>>>(
      xbf, wout_bf, d_out, MROWS, DM, DM);
}
